// TRILIXAttention_34729105555790
// MI455X (gfx1250) — compile-verified, matrix operand order checked by reading, not run on hardware
//
#include <hip/hip_runtime.h>
#include <hip/hip_bf16.h>

// ---------------------------------------------------------------------------
// MI455X (gfx1250) GQA attention layer.
// - All matmuls via v_wmma_f32_16x16x32_f16 (K=32/issue vs K=4 for f32 WMMA),
//   fp32 accumulate, fp32 softmax (v_exp_f32 co-executes with XDL WMMA).
// - Natural-layout tiles (GEMM A, flash K) stream through the Tensor Data
//   Mover, double-buffered: tile i+1 is issued before computing tile i and
//   s_wait_tensorcnt(1) retires tile i (TDM ops from one wave are in-order).
// - Transposed tiles (GEMM B, flash V^T) are register-prefetched one
//   iteration ahead and scattered to LDS.
// NOTE: no pointer ARRAYS into __shared__ (ld.lld cannot relocate the
// addrspacecast aggregate initializer) — double-buffer pointers are computed
// arithmetically per iteration instead.
// ---------------------------------------------------------------------------

typedef _Float16 v16h __attribute__((ext_vector_type(16)));
typedef float    v8f  __attribute__((ext_vector_type(8)));
typedef unsigned int u32x4 __attribute__((ext_vector_type(4)));
typedef int          i32x4 __attribute__((ext_vector_type(4)));
typedef int          i32x8 __attribute__((ext_vector_type(8)));

union FragH {          // one 16x32 f16 WMMA operand = 8 VGPRs = 2x16 bytes/lane
    v16h v;
    uint4 q[2];
};

#define SEQ      2048
#define HIDDEN   2048
#define NHEADS   32
#define NKV      8
#define HDIM     64
#define GQA_G    (NHEADS / NKV)

// ---- Tensor Data Mover availability (device pass only; host pass falls back)
#define USE_TDM 0
#if defined(__HIP_DEVICE_COMPILE__)
#ifdef __has_builtin
#if __has_builtin(__builtin_amdgcn_tensor_load_to_lds)
#undef USE_TDM
#define USE_TDM 1
#endif
#endif
#endif

#if USE_TDM
// Build a D# for a 2-D tile of 2-byte elements and issue TENSOR_LOAD_TO_LDS.
// Group0: count=1 | lds_addr | global_addr | type=2.  Group1: data_size=2B,
// tensor_dim0/1, tile_dim0/1, tensor_dim0_stride.  Groups 2/3 zero (<=2D).
__device__ __forceinline__ void tdm_load_tile_2d(const void* gptr, unsigned lds_byte_off,
                                                 unsigned tdim0, unsigned tdim1,
                                                 unsigned tile0, unsigned tile1,
                                                 unsigned stride0_elems) {
    unsigned long long ga = (unsigned long long)gptr;
    u32x4 g0;
    g0[0] = 1u;                                                  // count=1 (valid)
    g0[1] = lds_byte_off;                                        // lds_addr
    g0[2] = (unsigned)ga;                                        // global_addr[31:0]
    g0[3] = (unsigned)((ga >> 32) & 0x01ffffffu) | (2u << 30);   // [56:32] | type=2
    i32x8 g1;
    g1[0] = (int)(1u << 16);                                     // mask=0, data_size=2B
    g1[1] = (int)((tdim0 & 0xffffu) << 16);                      // tensor_dim0[15:0]
    g1[2] = (int)(((tdim0 >> 16) & 0xffffu) | ((tdim1 & 0xffffu) << 16));
    g1[3] = (int)(((tdim1 >> 16) & 0xffffu) | ((tile0 & 0xffffu) << 16));
    g1[4] = (int)(tile1 & 0xffffu);                              // tile_dim1, tile_dim2=0
    g1[5] = (int)stride0_elems;                                  // dim0_stride[31:0]
    g1[6] = 0;                                                   // stride hi, dim1_stride lo
    g1[7] = 0;
    i32x4 z4 = {0, 0, 0, 0};
#if __clang_major__ >= 23
    i32x8 z8 = {0, 0, 0, 0, 0, 0, 0, 0};
    __builtin_amdgcn_tensor_load_to_lds(g0, g1, z4, z4, z8, 0);
#else
    __builtin_amdgcn_tensor_load_to_lds(g0, g1, z4, z4, 0);
#endif
}
#endif

// ---------------------------------------------------------------------------
// f32 -> f16 conversion (vectorized by 4)
// ---------------------------------------------------------------------------
__global__ __launch_bounds__(256) void cvt_f32_f16(const float* __restrict__ src,
                                                   _Float16* __restrict__ dst, int n4) {
    int i = blockIdx.x * blockDim.x + threadIdx.x;
    if (i >= n4) return;
    float4 f = reinterpret_cast<const float4*>(src)[i];
    _Float16 h0 = (_Float16)f.x, h1 = (_Float16)f.y, h2 = (_Float16)f.z, h3 = (_Float16)f.w;
    ushort4 packed;
    packed.x = *(unsigned short*)&h0;
    packed.y = *(unsigned short*)&h1;
    packed.z = *(unsigned short*)&h2;
    packed.w = *(unsigned short*)&h3;
    reinterpret_cast<ushort4*>(dst)[i] = packed;
}

// ---------------------------------------------------------------------------
// Tiled f16 GEMM: C(f32, MxN) = A(f16, MxK) * B(f16, KxN), row-major.
// 256 threads = 8 waves, block tile 128x128, K-step 32, double-buffered LDS.
// A tiles stream through the TDM (issue i+1, wait tensorcnt<=1 -> tile i
// resident, DMA of i+1 overlaps the 8 WMMAs of i).  B rows are register-
// prefetched one iteration ahead, then transpose-scattered to [n][k] LDS.
// ---------------------------------------------------------------------------
template <int N, int K>
__global__ __launch_bounds__(256) void gemm_f16_wmma(const _Float16* __restrict__ A,
                                                     const _Float16* __restrict__ B,
                                                     float* __restrict__ C, int M) {
    // [0,16K): A tiles (2 x 8KB, TDM targets), [16K,32K): B tiles (2 x 8KB)
    __shared__ __align__(16) unsigned char smem[32 * 1024];

    const int m0 = blockIdx.y * 128;
    const int n0 = blockIdx.x * 128;
    const int t    = threadIdx.x;
    const int wave = t >> 5;
    const int lane = t & 31;
    const int half = lane >> 4;     // which K-chunk this lane holds
    const int r    = lane & 15;     // row-in-tile / col-in-tile

    constexpr int NIT = K / 32;
    const int bk  = t >> 3;          // B staging: K-row
    const int bn  = (t & 7) * 16;    // B staging: N-col base
    const _Float16* bbase = B + (size_t)bk * N + n0 + bn;

    v8f acc[8];
    #pragma unroll
    for (int i = 0; i < 8; ++i) acc[i] = {};

    // ---- prologue: B rows for tile 0 -> regs; A tile 0 -> LDS buf 0 ----
    FragH brow;
    brow.q[0] = *reinterpret_cast<const uint4*>(bbase);
    brow.q[1] = *reinterpret_cast<const uint4*>(bbase + 8);
#if USE_TDM
    if (t < 32)
        tdm_load_tile_2d(A + (size_t)m0 * K, 0u, (unsigned)K, (unsigned)M,
                         32u, 128u, (unsigned)K);
#else
    const int ar = t >> 1;
    const int ak = (t & 1) * 16;
    uint4 a0 = *reinterpret_cast<const uint4*>(A + (size_t)(m0 + ar) * K + ak);
    uint4 a1 = *reinterpret_cast<const uint4*>(A + (size_t)(m0 + ar) * K + ak + 8);
#endif

    for (int i = 0; i < NIT; ++i) {
        const int cur = i & 1;
        // double-buffer pointers via arithmetic (no LDS pointer aggregates)
        _Float16* la = (_Float16*)(smem + cur * 8192);
        _Float16* lb = (_Float16*)(smem + 16384 + cur * 8192);

#if USE_TDM
        // Issue next A tile into the other buffer, retire the current one.
        if (i + 1 < NIT) {
            if (t < 32) {
                tdm_load_tile_2d(A + (size_t)m0 * K + (i + 1) * 32,
                                 cur ? 0u : 8192u, (unsigned)K, (unsigned)M,
                                 32u, 128u, (unsigned)K);
                __builtin_amdgcn_s_wait_tensorcnt(1);
            }
        } else {
            if (t < 32) __builtin_amdgcn_s_wait_tensorcnt(0);
        }
#else
        *reinterpret_cast<uint4*>(&la[ar * 32 + ak])     = a0;
        *reinterpret_cast<uint4*>(&la[ar * 32 + ak + 8]) = a1;
#endif
        // scatter current B rows to [n][k]; prefetch next B rows into regs
        #pragma unroll
        for (int e = 0; e < 16; ++e)
            lb[(bn + e) * 32 + bk] = brow.v[e];

        if (i + 1 < NIT) {
            const _Float16* bgp = bbase + (size_t)(i + 1) * 32 * N;
            __builtin_prefetch(bgp + 32 * N, 0, 1);   // global_prefetch_b8: tile i+2
            brow.q[0] = *reinterpret_cast<const uint4*>(bgp);
            brow.q[1] = *reinterpret_cast<const uint4*>(bgp + 8);
#if !USE_TDM
            a0 = *reinterpret_cast<const uint4*>(A + (size_t)(m0 + ar) * K + (i + 1) * 32 + ak);
            a1 = *reinterpret_cast<const uint4*>(A + (size_t)(m0 + ar) * K + (i + 1) * 32 + ak + 8);
#endif
        }
        __syncthreads();   // publish la, lb

        // ---- A fragment for this wave's 16 rows ----
        FragH af;
        const _Float16* ap = &la[(wave * 16 + r) * 32 + half * 8];
        af.q[0] = *reinterpret_cast<const uint4*>(ap);
        af.q[1] = *reinterpret_cast<const uint4*>(ap + 16);

        // ---- sweep 8 N-tiles (A-frag register reuse) ----
        #pragma unroll
        for (int nt = 0; nt < 8; ++nt) {
            FragH bf;
            const _Float16* bp = &lb[(nt * 16 + r) * 32 + half * 8];
            bf.q[0] = *reinterpret_cast<const uint4*>(bp);
            bf.q[1] = *reinterpret_cast<const uint4*>(bp + 16);
            acc[nt] = __builtin_amdgcn_wmma_f32_16x16x32_f16(
                false, af.v, false, bf.v, (short)0, acc[nt], false, false);
        }
        __syncthreads();   // all waves done with [cur] before it is re-staged
    }

    // ---- epilogue: C layout (row = j + half*8, col = r), constant offsets ----
    float* cp = C + (size_t)(m0 + wave * 16 + half * 8) * N + n0 + r;
    #pragma unroll
    for (int j = 0; j < 8; ++j) {
        #pragma unroll
        for (int nt = 0; nt < 8; ++nt) {
            cp[(size_t)j * N + nt * 16] = acc[nt][j];
        }
    }
}

// ---------------------------------------------------------------------------
// RoPE + [S][nh*64] f32 -> [nh][S][64] f16 head transpose.
// ---------------------------------------------------------------------------
__global__ __launch_bounds__(256) void rope_transpose(const float* __restrict__ src,
                                                      _Float16* __restrict__ dst,
                                                      int nheads) {
    int idx = blockIdx.x * blockDim.x + threadIdx.x;
    int total = nheads * SEQ * (HDIM / 2);
    if (idx >= total) return;
    int i = idx & 31;                 // pair index 0..31
    int s = (idx >> 5) & (SEQ - 1);
    int h = idx >> 16;                // 5 + 11 bits

    const float* xp = src + (size_t)s * (nheads * HDIM) + h * HDIM + 2 * i;
    float x1 = xp[0], x2 = xp[1];
    float tpos = (float)s;
    int d0 = 2 * i, d1 = 2 * i + 1;
    float f0 = __powf(500000.0f, -(float)(d0 & 31) / 32.0f);
    float f1 = __powf(500000.0f, -(float)(d1 & 31) / 32.0f);
    float c0 = __cosf(tpos * f0), s0 = __sinf(tpos * f0);
    float c1 = __cosf(tpos * f1), s1 = __sinf(tpos * f1);
    float o0 = x1 * c0 - x2 * s0;
    float o1 = x2 * c1 + x1 * s1;

    _Float16* dp = dst + ((size_t)h * SEQ + s) * HDIM + 2 * i;
    dp[0] = (_Float16)o0;
    dp[1] = (_Float16)o1;
}

// ---------------------------------------------------------------------------
// V: [S][nkv*64] f32 -> [nkv][S][64] f16
// ---------------------------------------------------------------------------
__global__ __launch_bounds__(256) void v_transpose(const float* __restrict__ src,
                                                   _Float16* __restrict__ dst) {
    int idx = blockIdx.x * blockDim.x + threadIdx.x;
    int total = SEQ * NKV * HDIM;
    if (idx >= total) return;
    int c = idx % (NKV * HDIM);
    int s = idx / (NKV * HDIM);
    int h = c / HDIM, d = c % HDIM;
    dst[((size_t)h * SEQ + s) * HDIM + d] = (_Float16)src[idx];
}

// ---------------------------------------------------------------------------
// Flash attention (causal, online softmax), f16 WMMA, fp32 softmax.
// Grid: (NHEADS, SEQ/64). Block: 128 threads = 4 waves, 16 q-rows/wave.
// K tiles stream via TDM (double-buffered, pipelined); V^T double-buffered
// with register prefetch.  Two barriers per key tile: publish (K,V^T) and
// publish P (the latter also fences the K-buffer WAR for the TDM write).
// ---------------------------------------------------------------------------
__global__ __launch_bounds__(128) void flash_attn(const _Float16* __restrict__ qh,
                                                  const _Float16* __restrict__ kh,
                                                  const _Float16* __restrict__ vh,
                                                  _Float16* __restrict__ attn_out) {
    // [0,8K): K tiles (2 x 4KB, TDM targets), [8K,16K): V^T (2 x 4KB),
    // [16K,20K): per-wave P tiles
    __shared__ __align__(16) unsigned char smem[20 * 1024];

    const int h    = blockIdx.x;
    const int qblk = blockIdx.y;
    const int hkv  = h / GQA_G;
    const int t    = threadIdx.x;
    const int wave = t >> 5;
    const int lane = t & 31;
    const int half = lane >> 4;
    const int r    = lane & 15;
    const int qb   = qblk * 64 + wave * 16;     // this wave's first q row
    _Float16* my_p = (_Float16*)(smem + 16384) + wave * (16 * 32);

    // Q fragments for both 32-wide D-chunks (constant across the key loop)
    FragH qf[2];
    #pragma unroll
    for (int dc = 0; dc < 2; ++dc) {
        const _Float16* qp = qh + ((size_t)h * SEQ + qb + r) * HDIM + dc * 32 + half * 8;
        qf[dc].q[0] = *reinterpret_cast<const uint4*>(qp);
        qf[dc].q[1] = *reinterpret_cast<const uint4*>(qp + 16);
    }

    v8f o[4];
    #pragma unroll
    for (int i = 0; i < 4; ++i) o[i] = {};
    float mrow[8], lrow[8];
    #pragma unroll
    for (int j = 0; j < 8; ++j) { mrow[j] = -1e30f; lrow[j] = 0.0f; }

    const float scale = 0.125f;               // 1/sqrt(64)
    const int ktiles = qblk * 2 + 2;          // uniform per block (causal bound)

    const int skey  = t >> 2;                 // staging: key row
    const int sdoff = (t & 3) * 16;           // staging: d offset
    const _Float16* vbase  = vh + ((size_t)hkv * SEQ + skey) * HDIM + sdoff;
#if !USE_TDM
    const _Float16* kbase  = kh + ((size_t)hkv * SEQ + skey) * HDIM + sdoff;
#endif

    // ---- prologue: V rows for tile 0 -> regs; K tile 0 -> LDS buf 0 ----
    FragH vrow;
    vrow.q[0] = *reinterpret_cast<const uint4*>(vbase);
    vrow.q[1] = *reinterpret_cast<const uint4*>(vbase + 8);
#if USE_TDM
    if (t < 32)
        tdm_load_tile_2d(kh + (size_t)hkv * SEQ * HDIM, 0u,
                         (unsigned)HDIM, (unsigned)SEQ, (unsigned)HDIM, 32u,
                         (unsigned)HDIM);
#else
    uint4 k0 = *reinterpret_cast<const uint4*>(kbase);
    uint4 k1 = *reinterpret_cast<const uint4*>(kbase + 8);
#endif

    for (int kt = 0; kt < ktiles; ++kt) {
        const int cur = kt & 1;
        _Float16* lk  = (_Float16*)(smem + cur * 4096);
        _Float16* lvt = (_Float16*)(smem + 8192 + cur * 4096);

#if USE_TDM
        if (kt + 1 < ktiles) {
            if (t < 32) {
                tdm_load_tile_2d(kh + ((size_t)hkv * SEQ + (kt + 1) * 32) * HDIM,
                                 cur ? 0u : 4096u,
                                 (unsigned)HDIM, (unsigned)SEQ, (unsigned)HDIM, 32u,
                                 (unsigned)HDIM);
                __builtin_amdgcn_s_wait_tensorcnt(1);
            }
        } else {
            if (t < 32) __builtin_amdgcn_s_wait_tensorcnt(0);
        }
#else
        *reinterpret_cast<uint4*>(&lk[skey * HDIM + sdoff])     = k0;
        *reinterpret_cast<uint4*>(&lk[skey * HDIM + sdoff + 8]) = k1;
#endif
        // scatter current V rows -> V^T; prefetch next V (and K) rows
        #pragma unroll
        for (int e = 0; e < 16; ++e)
            lvt[(sdoff + e) * 32 + skey] = vrow.v[e];

        if (kt + 1 < ktiles) {
            const _Float16* vgp = vbase + (size_t)(kt + 1) * 32 * HDIM;
            vrow.q[0] = *reinterpret_cast<const uint4*>(vgp);
            vrow.q[1] = *reinterpret_cast<const uint4*>(vgp + 8);
#if !USE_TDM
            const _Float16* kgp = kbase + (size_t)(kt + 1) * 32 * HDIM;
            k0 = *reinterpret_cast<const uint4*>(kgp);
            k1 = *reinterpret_cast<const uint4*>(kgp + 8);
#endif
        }
        __syncthreads();   // publish lk, lvt

        // ---- scores: two 16-key subtiles, accumulate over two D-chunks ----
        v8f sacc[2];
        #pragma unroll
        for (int st = 0; st < 2; ++st) {
            sacc[st] = {};
            #pragma unroll
            for (int dc = 0; dc < 2; ++dc) {
                FragH kf;
                const _Float16* kp = &lk[(st * 16 + r) * HDIM + dc * 32 + half * 8];
                kf.q[0] = *reinterpret_cast<const uint4*>(kp);
                kf.q[1] = *reinterpret_cast<const uint4*>(kp + 16);
                sacc[st] = __builtin_amdgcn_wmma_f32_16x16x32_f16(
                    false, qf[dc].v, false, kf.v, (short)0, sacc[st], false, false);
            }
        }

        // ---- online softmax on C-layout (row = j + half*8, col = r) ----
        const int col0 = kt * 32 + r;
        const int col1 = col0 + 16;
        #pragma unroll
        for (int j = 0; j < 8; ++j) {
            const int row = qb + j + half * 8;
            float s0 = (col0 <= row) ? sacc[0][j] * scale : -1e30f;
            float s1 = (col1 <= row) ? sacc[1][j] * scale : -1e30f;

            float tmax = fmaxf(s0, s1);
            #pragma unroll
            for (int off = 1; off < 16; off <<= 1)
                tmax = fmaxf(tmax, __shfl_xor(tmax, off, 32));   // stays in half

            float mnew  = fmaxf(mrow[j], tmax);
            float corr  = __expf(mrow[j] - mnew);
            float p0    = __expf(s0 - mnew);
            float p1    = __expf(s1 - mnew);
            float psum  = p0 + p1;
            #pragma unroll
            for (int off = 1; off < 16; off <<= 1)
                psum += __shfl_xor(psum, off, 32);

            lrow[j] = lrow[j] * corr + psum;
            mrow[j] = mnew;
            #pragma unroll
            for (int t4 = 0; t4 < 4; ++t4) o[t4][j] *= corr;

            my_p[(j + half * 8) * 32 + r]      = (_Float16)p0;
            my_p[(j + half * 8) * 32 + 16 + r] = (_Float16)p1;
        }
        __syncthreads();   // publish P; also fences scores' K-buffer reads

        // ---- PV: A = P (16x32 keys), B = V^T chunks (4 output d-tiles) ----
        FragH pf;
        const _Float16* pp = &my_p[r * 32 + half * 8];
        pf.q[0] = *reinterpret_cast<const uint4*>(pp);
        pf.q[1] = *reinterpret_cast<const uint4*>(pp + 16);
        #pragma unroll
        for (int t4 = 0; t4 < 4; ++t4) {
            FragH vf;
            const _Float16* vp = &lvt[(t4 * 16 + r) * 32 + half * 8];
            vf.q[0] = *reinterpret_cast<const uint4*>(vp);
            vf.q[1] = *reinterpret_cast<const uint4*>(vp + 16);
            o[t4] = __builtin_amdgcn_wmma_f32_16x16x32_f16(
                false, pf.v, false, vf.v, (short)0, o[t4], false, false);
        }
    }

    // ---- normalize and write [S][NHEADS*HDIM] f16 for the O-projection ----
    #pragma unroll
    for (int j = 0; j < 8; ++j) {
        float inv_l = 1.0f / lrow[j];
        const size_t row = (size_t)qb + j + half * 8;
        #pragma unroll
        for (int t4 = 0; t4 < 4; ++t4) {
            attn_out[row * (NHEADS * HDIM) + h * HDIM + t4 * 16 + r] =
                (_Float16)(o[t4][j] * inv_l);
        }
    }
}

// ---------------------------------------------------------------------------
// Host-side orchestration
// ---------------------------------------------------------------------------
extern "C" void kernel_launch(void* const* d_in, const int* in_sizes, int n_in,
                              void* d_out, int out_size, void* d_ws, size_t ws_size,
                              hipStream_t stream) {
    const float* hidden = (const float*)d_in[0];
    // d_in[1] = attention_mask: causal mask recomputed on device, not needed
    const float* Wq = (const float*)d_in[2];
    const float* Wk = (const float*)d_in[3];
    const float* Wv = (const float*)d_in[4];
    const float* Wo = (const float*)d_in[5];
    float* out = (float*)d_out;

    char* ws = (char*)d_ws;
    size_t off = 0;
    auto alloc = [&](size_t bytes) { char* p = ws + off; off += (bytes + 255) & ~size_t(255); return p; };

    _Float16* hs16   = (_Float16*)alloc((size_t)SEQ * HIDDEN * 2);
    _Float16* wq16   = (_Float16*)alloc((size_t)HIDDEN * NHEADS * HDIM * 2);
    _Float16* wk16   = (_Float16*)alloc((size_t)HIDDEN * NKV * HDIM * 2);
    _Float16* wv16   = (_Float16*)alloc((size_t)HIDDEN * NKV * HDIM * 2);
    _Float16* wo16   = (_Float16*)alloc((size_t)NHEADS * HDIM * HIDDEN * 2);
    float*    qf     = (float*)   alloc((size_t)SEQ * NHEADS * HDIM * 4);
    float*    kf     = (float*)   alloc((size_t)SEQ * NKV * HDIM * 4);
    float*    vf     = (float*)   alloc((size_t)SEQ * NKV * HDIM * 4);
    _Float16* qhp    = (_Float16*)alloc((size_t)NHEADS * SEQ * HDIM * 2);
    _Float16* khp    = (_Float16*)alloc((size_t)NKV * SEQ * HDIM * 2);
    _Float16* vhp    = (_Float16*)alloc((size_t)NKV * SEQ * HDIM * 2);
    _Float16* attn16 = (_Float16*)alloc((size_t)SEQ * NHEADS * HDIM * 2);

    // 1) downconvert activations + weights to f16
    {
        int n4;
        n4 = SEQ * HIDDEN / 4;
        cvt_f32_f16<<<(n4 + 255) / 256, 256, 0, stream>>>(hidden, hs16, n4);
        n4 = HIDDEN * NHEADS * HDIM / 4;
        cvt_f32_f16<<<(n4 + 255) / 256, 256, 0, stream>>>(Wq, wq16, n4);
        n4 = HIDDEN * NKV * HDIM / 4;
        cvt_f32_f16<<<(n4 + 255) / 256, 256, 0, stream>>>(Wk, wk16, n4);
        cvt_f32_f16<<<(n4 + 255) / 256, 256, 0, stream>>>(Wv, wv16, n4);
        n4 = NHEADS * HDIM * HIDDEN / 4;
        cvt_f32_f16<<<(n4 + 255) / 256, 256, 0, stream>>>(Wo, wo16, n4);
    }

    // 2) QKV projections (f16 WMMA, f32 accum out)
    gemm_f16_wmma<NHEADS * HDIM, HIDDEN>
        <<<dim3((NHEADS * HDIM) / 128, SEQ / 128), 256, 0, stream>>>(hs16, wq16, qf, SEQ);
    gemm_f16_wmma<NKV * HDIM, HIDDEN>
        <<<dim3((NKV * HDIM) / 128, SEQ / 128), 256, 0, stream>>>(hs16, wk16, kf, SEQ);
    gemm_f16_wmma<NKV * HDIM, HIDDEN>
        <<<dim3((NKV * HDIM) / 128, SEQ / 128), 256, 0, stream>>>(hs16, wv16, vf, SEQ);

    // 3) RoPE + head transpose; V transpose
    {
        int n = NHEADS * SEQ * (HDIM / 2);
        rope_transpose<<<(n + 255) / 256, 256, 0, stream>>>(qf, qhp, NHEADS);
        n = NKV * SEQ * (HDIM / 2);
        rope_transpose<<<(n + 255) / 256, 256, 0, stream>>>(kf, khp, NKV);
        n = SEQ * NKV * HDIM;
        v_transpose<<<(n + 255) / 256, 256, 0, stream>>>(vf, vhp);
    }

    // 4) fused causal flash attention
    flash_attn<<<dim3(NHEADS, SEQ / 64), 128, 0, stream>>>(qhp, khp, vhp, attn16);

    // 5) output projection -> fp32 result
    gemm_f16_wmma<HIDDEN, HIDDEN>
        <<<dim3(HIDDEN / 128, SEQ / 128), 256, 0, stream>>>(attn16, wo16, out, SEQ);
}
